// MoE_790273983069
// MI455X (gfx1250) — compile-verified, matrix operand order checked by reading, not run on hardware
//
#include <hip/hip_runtime.h>

// ---------------- problem constants ----------------
#define B_TOK  8192
#define D_IN   1024
#define D_OUT  1024
#define HDIM   2048
#define NEXP   8
#define KTOP   4
#define NTASK  10
#define TDIM   512
#define TDIM2  64
#define NSLOTS (B_TOK * KTOP)   // each token routed to exactly KTOP experts

typedef __attribute__((ext_vector_type(16))) _Float16 v16h;
typedef __attribute__((ext_vector_type(8)))  _Float16 v8h;
typedef __attribute__((ext_vector_type(8)))  float    v8f;

// ---------------- gfx1250 async copy path (guarded; fallback = sync copy) -------
#if defined(__AMDGCN__) && __has_builtin(__builtin_amdgcn_global_load_async_to_lds_b128) && \
    __has_builtin(__builtin_amdgcn_s_wait_asynccnt)
#define MOE_ASYNC 1
#else
#define MOE_ASYNC 0
#endif

typedef int v4i __attribute__((vector_size(16)));
#if MOE_ASYNC
typedef __attribute__((address_space(1))) v4i gv4i;   // global
typedef __attribute__((address_space(3))) v4i lv4i;   // LDS
#endif

__device__ __forceinline__ void cp16(const _Float16* g, _Float16* l) {
#if MOE_ASYNC
    __builtin_amdgcn_global_load_async_to_lds_b128((gv4i*)g, (lv4i*)l, 0, 0);
#else
    *(v8h*)l = *(const v8h*)g;
#endif
}

__device__ __forceinline__ void wait_cp() {
#if MOE_ASYNC
    __builtin_amdgcn_s_wait_asynccnt(0);
#endif
}

// ---------------- fp32 -> fp16 convert ----------------
__global__ void cvt_kernel(const float* __restrict__ src, _Float16* __restrict__ dst, int n) {
    int i = blockIdx.x * blockDim.x + threadIdx.x;
    int stride = gridDim.x * blockDim.x;
    for (; i < n; i += stride) dst[i] = (_Float16)src[i];
}

// ------- fp32 [E][K][N] -> fp16 [E][N][K] transposed convert (tiled via LDS) -------
__global__ void cvt_t_kernel(const float* __restrict__ src, _Float16* __restrict__ dst,
                             int K, int N) {
    __shared__ _Float16 tile[32][33];
    src += (size_t)blockIdx.z * K * N;
    dst += (size_t)blockIdx.z * K * N;
    const int n0 = blockIdx.x * 32, k0 = blockIdx.y * 32;
    const int tx = threadIdx.x & 31, ty = threadIdx.x >> 5;   // 32 x 8
#pragma unroll
    for (int i = 0; i < 32; i += 8)
        tile[ty + i][tx] = (_Float16)src[(size_t)(k0 + ty + i) * N + n0 + tx];
    __syncthreads();
#pragma unroll
    for (int i = 0; i < 32; i += 8)
        dst[(size_t)(n0 + ty + i) * K + k0 + tx] = tile[tx][ty + i];
}

// ---------------- task embedding table: [NTASK, TDIM2] ----------------
__global__ void temb_kernel(const float* __restrict__ tm, const float* __restrict__ tW,
                            const float* __restrict__ tb, float* __restrict__ tembT) {
    int i = blockIdx.x * blockDim.x + threadIdx.x;
    if (i >= NTASK * TDIM2) return;
    int t = i / TDIM2, j = i % TDIM2;
    float acc = tb[j];
    for (int k = 0; k < TDIM; k++)
        acc += tm[(size_t)t * TDIM + k] * tW[(size_t)k * TDIM2 + j];
    tembT[i] = acc;
}

// ---------------- gating: logits + top-4 softmax -> gates[B, NEXP] ----------------
__global__ void gate_kernel(const float* __restrict__ x, const int* __restrict__ task_ids,
                            const float* __restrict__ tembT, const float* __restrict__ w_gate,
                            float* __restrict__ gates) {
    int b = blockIdx.x * blockDim.x + threadIdx.x;
    if (b >= B_TOK) return;
    float lg[NEXP];
#pragma unroll
    for (int e = 0; e < NEXP; e++) lg[e] = 0.f;
    const float* xr = x + (size_t)b * D_IN;
    for (int k = 0; k < D_IN; k++) {
        float xv = xr[k];
        const float* wr = w_gate + (size_t)k * NEXP;
#pragma unroll
        for (int e = 0; e < NEXP; e++) lg[e] += xv * wr[e];
    }
    int t = task_ids[b];
    const float* te = tembT + (size_t)t * TDIM2;
    for (int k = 0; k < TDIM2; k++) {
        float tv = te[k];
        const float* wr = w_gate + (size_t)(D_IN + k) * NEXP;
#pragma unroll
        for (int e = 0; e < NEXP; e++) lg[e] += tv * wr[e];
    }
    // exact top-4 (descending, ties -> lower index first, as lax.top_k)
    bool used[NEXP];
#pragma unroll
    for (int e = 0; e < NEXP; e++) used[e] = false;
    int   idx[KTOP];
    float val[KTOP];
    for (int k = 0; k < KTOP; k++) {
        int best = 0; float bv = -3.4e38f;
#pragma unroll
        for (int e = 0; e < NEXP; e++)
            if (!used[e] && lg[e] > bv) { bv = lg[e]; best = e; }
        used[best] = true; idx[k] = best; val[k] = bv;
    }
    float m = val[0];
    float s = 0.f, ex[KTOP];
#pragma unroll
    for (int k = 0; k < KTOP; k++) { ex[k] = __expf(val[k] - m); s += ex[k]; }
    float out[NEXP];
#pragma unroll
    for (int e = 0; e < NEXP; e++) out[e] = 0.f;
#pragma unroll
    for (int k = 0; k < KTOP; k++) out[idx[k]] = ex[k] / s;
#pragma unroll
    for (int e = 0; e < NEXP; e++) gates[(size_t)b * NEXP + e] = out[e];
}

// ---------------- routing: stable compaction + load-balance loss ----------------
__global__ void route_kernel(const float* __restrict__ gates,
                             int* __restrict__ route, int* __restrict__ slotFor,
                             int* __restrict__ counts, int* __restrict__ offsets,
                             float* __restrict__ loss_out) {
    __shared__ float s_imp[NEXP];
    __shared__ int   s_cnt[NEXP];
    __shared__ int   s_off[NEXP + 1];
    __shared__ float s_part[256];

    const int tid  = threadIdx.x;
    const int lane = tid & 31;
    const int w    = tid >> 5;   // expert id

    float imp = 0.f;
    int   cnt = 0;
    for (int base = 0; base < B_TOK; base += 32) {
        float g = gates[(size_t)(base + lane) * NEXP + w];
        unsigned m = __builtin_amdgcn_ballot_w32(g > 0.f);
        cnt += __popc(m);
        imp += g;
    }
    s_part[tid] = imp;
    __syncthreads();
    if (lane == 0) {
        float s = 0.f;
        for (int i = 0; i < 32; i++) s += s_part[w * 32 + i];
        s_imp[w] = s;
        s_cnt[w] = cnt;
    }
    __syncthreads();
    if (tid == 0) {
        int o = 0;
        for (int e = 0; e < NEXP; e++) { s_off[e] = o; o += s_cnt[e]; counts[e] = s_cnt[e]; }
        s_off[NEXP] = o;
        for (int e = 0; e <= NEXP; e++) offsets[e] = s_off[e];
        float mi = 0.f, ml = 0.f;
        for (int e = 0; e < NEXP; e++) { mi += s_imp[e]; ml += (float)s_cnt[e]; }
        mi /= NEXP; ml /= NEXP;
        float vi = 0.f, vl = 0.f;
        for (int e = 0; e < NEXP; e++) {
            float di = s_imp[e] - mi; vi += di * di;
            float dl = (float)s_cnt[e] - ml; vl += dl * dl;
        }
        vi /= (NEXP - 1); vl /= (NEXP - 1);
        loss_out[0] = vi / (mi * mi + 1e-10f) + vl / (ml * ml + 1e-10f);
    }
    __syncthreads();
    // pass 2: stable write of route list + token->slot map
    int pos = 0;
    for (int base = 0; base < B_TOK; base += 32) {
        float g = gates[(size_t)(base + lane) * NEXP + w];
        unsigned m = __builtin_amdgcn_ballot_w32(g > 0.f);
        int rank = __popc(m & ((1u << lane) - 1u));
        if (g > 0.f) {
            int slot = s_off[w] + pos + rank;
            route[slot] = base + lane;
            slotFor[(size_t)(base + lane) * NEXP + w] = slot;
        }
        pos += __popc(m);
    }
}

// ---------------- GEMM1: h[slot,:] = relu(x[route[slot]] @ W1[e] + b1) ----------------
// 128x128 tile, 8 waves as 4(M)x2(N), K-step 32, double-buffered LDS + async copies.
// W1 is pre-transposed to [E][HDIM][D_IN] so A and B tiles are both contiguous copies.
__launch_bounds__(256)
__global__ void gemm1_kernel(const _Float16* __restrict__ xb,
                             const _Float16* __restrict__ w1t,
                             const float* __restrict__ fc1_b,
                             const int* __restrict__ route,
                             const int* __restrict__ counts,
                             const int* __restrict__ offsets,
                             _Float16* __restrict__ hbuf) {
    const int e  = blockIdx.x >> 6;
    const int mt = blockIdx.x & 63;
    const int nb = blockIdx.y * 128;
    const int cnt = counts[e];
    if (mt * 128 >= cnt) return;
    const int base = offsets[e];

    __shared__ __align__(16) _Float16 ldsA[2][128 * 32];   // [row][k]
    __shared__ __align__(16) _Float16 ldsB[2][128 * 32];   // [n][k]

    const int tid  = threadIdx.x;
    const int lane = tid & 31;
    const int wv   = tid >> 5;
    const int wM   = wv & 3;
    const int wN   = wv >> 2;
    const int kb   = (lane < 16) ? 0 : 8;

    // per-thread copy descriptors: 2 chunks of 8 halfs for A and B each
    const _Float16* gA[2]; int lA[2];
    const _Float16* gB[2]; int lB[2];
    const _Float16* wBt = w1t + (size_t)e * D_IN * HDIM;
#pragma unroll
    for (int c = 0; c < 2; c++) {
        int s = tid + c * 256;
        int row = s >> 2, c8 = (s & 3) * 8;
        int rA = mt * 128 + row; if (rA >= cnt) rA = cnt - 1;   // clamp: branch-free gather
        gA[c] = xb + (size_t)route[base + rA] * D_IN + c8;
        lA[c] = row * 32 + c8;
        gB[c] = wBt + (size_t)(nb + row) * D_IN + c8;
        lB[c] = row * 32 + c8;
    }

    v8f acc[2][4] = {};

    // prologue: fill buffer 0
#pragma unroll
    for (int c = 0; c < 2; c++) { cp16(gA[c], &ldsA[0][lA[c]]); cp16(gB[c], &ldsB[0][lB[c]]); }

    for (int kk = 0; kk < D_IN; kk += 32) {
        const int p = (kk >> 5) & 1;
        wait_cp();
        __syncthreads();
        if (kk + 32 < D_IN) {
#pragma unroll
            for (int c = 0; c < 2; c++) {
                cp16(gA[c] + kk + 32, &ldsA[p ^ 1][lA[c]]);
                cp16(gB[c] + kk + 32, &ldsB[p ^ 1][lB[c]]);
            }
        }
        v16h afr[2], bfr[4];
#pragma unroll
        for (int mi = 0; mi < 2; mi++) {
            int m = wM * 32 + mi * 16 + (lane & 15);
            v8h lo = *(const v8h*)(&ldsA[p][m * 32 + kb]);
            v8h hi = *(const v8h*)(&ldsA[p][m * 32 + kb + 16]);
            afr[mi] = __builtin_shufflevector(lo, hi, 0,1,2,3,4,5,6,7,8,9,10,11,12,13,14,15);
        }
#pragma unroll
        for (int ni = 0; ni < 4; ni++) {
            int n = wN * 64 + ni * 16 + (lane & 15);
            v8h lo = *(const v8h*)(&ldsB[p][n * 32 + kb]);
            v8h hi = *(const v8h*)(&ldsB[p][n * 32 + kb + 16]);
            bfr[ni] = __builtin_shufflevector(lo, hi, 0,1,2,3,4,5,6,7,8,9,10,11,12,13,14,15);
        }
#pragma unroll
        for (int mi = 0; mi < 2; mi++)
#pragma unroll
            for (int ni = 0; ni < 4; ni++)
                acc[mi][ni] = __builtin_amdgcn_wmma_f32_16x16x32_f16(
                    false, afr[mi], false, bfr[ni], (short)0, acc[mi][ni], false, false);
    }

    // epilogue: bias + relu + fp16 store
#pragma unroll
    for (int mi = 0; mi < 2; mi++) {
#pragma unroll
        for (int ni = 0; ni < 4; ni++) {
            int nc = nb + wN * 64 + ni * 16 + (lane & 15);
            float bias = fc1_b[e * HDIM + nc];
            v8f c = acc[mi][ni];
#pragma unroll
            for (int r = 0; r < 8; r++) {
                int rr = wM * 32 + mi * 16 + r + ((lane < 16) ? 0 : 8);
                int s = mt * 128 + rr;
                if (s < cnt) {
                    float v = c[r] + bias;
                    v = v > 0.f ? v : 0.f;
                    hbuf[(size_t)(base + s) * HDIM + nc] = (_Float16)v;
                }
            }
        }
    }
}

// ---------------- GEMM2: partial[slot,:] = h[slot] @ W2[e] + b2 ----------------
// W2 pre-transposed to [E][D_OUT][HDIM].
__launch_bounds__(256)
__global__ void gemm2_kernel(const _Float16* __restrict__ hbuf,
                             const _Float16* __restrict__ w2t,
                             const float* __restrict__ fc2_b,
                             const int* __restrict__ counts,
                             const int* __restrict__ offsets,
                             float* __restrict__ partial) {
    const int e  = blockIdx.x >> 6;
    const int mt = blockIdx.x & 63;
    const int nb = blockIdx.y * 128;
    const int cnt = counts[e];
    if (mt * 128 >= cnt) return;
    const int base = offsets[e];

    __shared__ __align__(16) _Float16 ldsA[2][128 * 32];
    __shared__ __align__(16) _Float16 ldsB[2][128 * 32];

    const int tid  = threadIdx.x;
    const int lane = tid & 31;
    const int wv   = tid >> 5;
    const int wM   = wv & 3;
    const int wN   = wv >> 2;
    const int kb   = (lane < 16) ? 0 : 8;

    const _Float16* gA[2]; int lA[2];
    const _Float16* gB[2]; int lB[2];
    const _Float16* wBt = w2t + (size_t)e * HDIM * D_OUT;
#pragma unroll
    for (int c = 0; c < 2; c++) {
        int s = tid + c * 256;
        int row = s >> 2, c8 = (s & 3) * 8;
        int rA = mt * 128 + row; if (rA >= cnt) rA = cnt - 1;
        gA[c] = hbuf + (size_t)(base + rA) * HDIM + c8;
        lA[c] = row * 32 + c8;
        gB[c] = wBt + (size_t)(nb + row) * HDIM + c8;
        lB[c] = row * 32 + c8;
    }

    v8f acc[2][4] = {};

#pragma unroll
    for (int c = 0; c < 2; c++) { cp16(gA[c], &ldsA[0][lA[c]]); cp16(gB[c], &ldsB[0][lB[c]]); }

    for (int kk = 0; kk < HDIM; kk += 32) {
        const int p = (kk >> 5) & 1;
        wait_cp();
        __syncthreads();
        if (kk + 32 < HDIM) {
#pragma unroll
            for (int c = 0; c < 2; c++) {
                cp16(gA[c] + kk + 32, &ldsA[p ^ 1][lA[c]]);
                cp16(gB[c] + kk + 32, &ldsB[p ^ 1][lB[c]]);
            }
        }
        v16h afr[2], bfr[4];
#pragma unroll
        for (int mi = 0; mi < 2; mi++) {
            int m = wM * 32 + mi * 16 + (lane & 15);
            v8h lo = *(const v8h*)(&ldsA[p][m * 32 + kb]);
            v8h hi = *(const v8h*)(&ldsA[p][m * 32 + kb + 16]);
            afr[mi] = __builtin_shufflevector(lo, hi, 0,1,2,3,4,5,6,7,8,9,10,11,12,13,14,15);
        }
#pragma unroll
        for (int ni = 0; ni < 4; ni++) {
            int n = wN * 64 + ni * 16 + (lane & 15);
            v8h lo = *(const v8h*)(&ldsB[p][n * 32 + kb]);
            v8h hi = *(const v8h*)(&ldsB[p][n * 32 + kb + 16]);
            bfr[ni] = __builtin_shufflevector(lo, hi, 0,1,2,3,4,5,6,7,8,9,10,11,12,13,14,15);
        }
#pragma unroll
        for (int mi = 0; mi < 2; mi++)
#pragma unroll
            for (int ni = 0; ni < 4; ni++)
                acc[mi][ni] = __builtin_amdgcn_wmma_f32_16x16x32_f16(
                    false, afr[mi], false, bfr[ni], (short)0, acc[mi][ni], false, false);
    }

#pragma unroll
    for (int mi = 0; mi < 2; mi++) {
#pragma unroll
        for (int ni = 0; ni < 4; ni++) {
            int nc = nb + wN * 64 + ni * 16 + (lane & 15);
            float bias = fc2_b[e * D_OUT + nc];
            v8f c = acc[mi][ni];
#pragma unroll
            for (int r = 0; r < 8; r++) {
                int rr = wM * 32 + mi * 16 + r + ((lane < 16) ? 0 : 8);
                int s = mt * 128 + rr;
                if (s < cnt)
                    partial[(size_t)(base + s) * D_OUT + nc] = c[r] + bias;
            }
        }
    }
}

// ---------------- combine: y = log( sum_e gate * exp(partial) ) ----------------
__global__ void combine_kernel(const float* __restrict__ gates, const int* __restrict__ slotFor,
                               const float* __restrict__ partial, float* __restrict__ y) {
    __shared__ float g[NEXP];
    __shared__ int   sl[NEXP];
    const int b = blockIdx.x;
    const int tid = threadIdx.x;
    if (tid < NEXP) {
        g[tid]  = gates[(size_t)b * NEXP + tid];
        sl[tid] = slotFor[(size_t)b * NEXP + tid];
    }
    __syncthreads();
    for (int o = tid; o < D_OUT; o += blockDim.x) {
        float acc = 0.f;
#pragma unroll
        for (int e = 0; e < NEXP; e++) {
            if (g[e] > 0.f)
                acc += g[e] * __expf(partial[(size_t)sl[e] * D_OUT + o]);
        }
        if (acc == 0.f) acc = 2.220446049250313e-16f;
        y[(size_t)b * D_OUT + o] = __logf(acc);
    }
}

// ---------------- host launcher ----------------
extern "C" void kernel_launch(void* const* d_in, const int* in_sizes, int n_in,
                              void* d_out, int out_size, void* d_ws, size_t ws_size,
                              hipStream_t stream) {
    (void)in_sizes; (void)n_in; (void)out_size; (void)ws_size;

    const float* x           = (const float*)d_in[0];
    const int*   task_ids    = (const int*)d_in[1];
    const float* task_matrix = (const float*)d_in[2];
    const float* task_W      = (const float*)d_in[3];
    const float* task_b      = (const float*)d_in[4];
    const float* w_gate      = (const float*)d_in[5];
    const float* fc1_w       = (const float*)d_in[6];
    const float* fc1_b       = (const float*)d_in[7];
    const float* fc2_w       = (const float*)d_in[8];
    const float* fc2_b       = (const float*)d_in[9];

    float* y    = (float*)d_out;
    float* loss = y + (size_t)B_TOK * D_OUT;

    char* ws = (char*)d_ws;
    size_t off = 0;
    auto alloc = [&](size_t bytes) -> void* {
        void* p = ws + off;
        off = (off + bytes + 255) & ~(size_t)255;
        return p;
    };
    float*    gates   = (float*)alloc((size_t)B_TOK * NEXP * 4);
    int*      slotFor = (int*)  alloc((size_t)B_TOK * NEXP * 4);
    int*      route   = (int*)  alloc((size_t)NSLOTS * 4);
    int*      counts  = (int*)  alloc(64);
    int*      offsets = (int*)  alloc(64);
    float*    tembT   = (float*)alloc((size_t)NTASK * TDIM2 * 4);
    _Float16* xb      = (_Float16*)alloc((size_t)B_TOK * D_IN * 2);
    _Float16* w1t     = (_Float16*)alloc((size_t)NEXP * D_IN * HDIM * 2);
    _Float16* w2t     = (_Float16*)alloc((size_t)NEXP * HDIM * D_OUT * 2);
    _Float16* hbuf    = (_Float16*)alloc((size_t)NSLOTS * HDIM * 2);
    float*    partial = (float*)alloc((size_t)NSLOTS * D_OUT * 4);

    cvt_kernel<<<4096, 256, 0, stream>>>(x, xb, B_TOK * D_IN);
    // W1: [E][K=1024][N=2048] -> [E][N][K];  W2: [E][K=2048][N=1024] -> [E][N][K]
    cvt_t_kernel<<<dim3(HDIM / 32, D_IN / 32, NEXP), 256, 0, stream>>>(fc1_w, w1t, D_IN, HDIM);
    cvt_t_kernel<<<dim3(D_OUT / 32, HDIM / 32, NEXP), 256, 0, stream>>>(fc2_w, w2t, HDIM, D_OUT);
    temb_kernel<<<3, 256, 0, stream>>>(task_matrix, task_W, task_b, tembT);
    gate_kernel<<<B_TOK / 256, 256, 0, stream>>>(x, task_ids, tembT, w_gate, gates);
    route_kernel<<<1, 256, 0, stream>>>(gates, route, slotFor, counts, offsets, loss);
    gemm1_kernel<<<dim3(NEXP * 64, HDIM / 128), 256, 0, stream>>>(xb, w1t, fc1_b, route,
                                                                  counts, offsets, hbuf);
    gemm2_kernel<<<dim3(NEXP * 64, D_OUT / 128), 256, 0, stream>>>(hbuf, w2t, fc2_b,
                                                                   counts, offsets, partial);
    combine_kernel<<<B_TOK, 256, 0, stream>>>(gates, slotFor, partial, y);
}
